// JambaMambaBlock_21577915695236
// MI455X (gfx1250) — matrix operand order repaired and checked by reading, not compiled, not run on hardware
//
#include <hip/hip_runtime.h>

#define T_TOK   4096
#define DMODEL  2048
#define DINNER  4096
#define DSTATE  128
#define NHEADS  64
#define HEADDIM 64
#define CONVDIM 4352
#define INPROJ  8512
#define INTERM  5632
#define CHUNKL  128
#define NCHUNK  16
#define LSEQ    2048
#define LDN     136   // padded row stride (elements) for SSD LDS tiles

typedef __attribute__((ext_vector_type(8)))  unsigned short u16x8;
typedef __attribute__((ext_vector_type(16))) unsigned short u16x16;
typedef __attribute__((ext_vector_type(16))) __bf16         v16bf;
typedef __attribute__((ext_vector_type(8)))  float          v8f;
typedef __attribute__((ext_vector_type(4)))  unsigned int   v4u;
typedef __attribute__((ext_vector_type(8)))  int            v8i;
typedef __attribute__((ext_vector_type(4)))  int            v4i;

#if defined(__HIP_DEVICE_COMPILE__) && __has_builtin(__builtin_amdgcn_tensor_load_to_lds) && __has_builtin(__builtin_amdgcn_s_wait_tensorcnt)
#define HAVE_TDM 1
#else
#define HAVE_TDM 0
#endif

__device__ __forceinline__ unsigned short f2bfu(float f) {
    unsigned u = __float_as_uint(f);
    u += 0x7fffu + ((u >> 16) & 1u);          // round-to-nearest-even
    return (unsigned short)(u >> 16);
}
__device__ __forceinline__ float bfu2f(unsigned short h) {
    return __uint_as_float(((unsigned)h) << 16);
}
__device__ __forceinline__ float silu_f(float x) { return x / (1.f + __expf(-x)); }

// ---- WMMA fragment builders (wave32; contiguous k = 16h + s, same order in A and B) -------------------------
// A (16-bit, 16x32): lanes 0-15 hold K=0..15, lanes 16-31 K=16..31
__device__ __forceinline__ v16bf frag_a(const unsigned short* row, int k0, int kh) {
    u16x8 lo = *(const u16x8*)(row + k0 + 16 * kh);
    u16x8 hi = *(const u16x8*)(row + k0 + 16 * kh + 8);
    u16x16 t;
#pragma unroll
    for (int j = 0; j < 8; ++j) { t[j] = lo[j]; t[j + 8] = hi[j]; }
    return __builtin_bit_cast(v16bf, t);
}
// B (16-bit, 32x16): lanes 0-15 hold K=0..15, lanes 16-31 K=16..31 (row = output col)
__device__ __forceinline__ v16bf frag_b(const unsigned short* row, int k0, int kh) {
    u16x8 lo = *(const u16x8*)(row + k0 + 16 * kh);
    u16x8 hi = *(const u16x8*)(row + k0 + 16 * kh + 8);
    u16x16 t;
#pragma unroll
    for (int j = 0; j < 8; ++j) { t[j] = lo[j]; t[j + 8] = hi[j]; }
    return __builtin_bit_cast(v16bf, t);
}
__device__ __forceinline__ v16bf frag_b_f32(const float* row, int k0, int kh) {
    u16x16 t;
#pragma unroll
    for (int j = 0; j < 16; ++j) t[j] = f2bfu(row[k0 + 16 * kh + j]);
    return __builtin_bit_cast(v16bf, t);
}
__device__ __forceinline__ v8f wmma_bf(v16bf a, v16bf b, v8f c) {
    return __builtin_amdgcn_wmma_f32_16x16x32_bf16(false, a, false, b, (short)0, c, false, false);
}

// ---- Tensor Data Mover: DMA one 128x32 bf16 tile (row stride K elements)
// from global into LDS with 16B padding after every 64B row => LDS row
// stride 40 elements, matching the fragment layout. Issued by one wave;
// completion tracked via TENSORcnt.
#if HAVE_TDM
__device__ __forceinline__ void tdm_load_A_tile(const unsigned short* gA, int M, int K,
                                                int m0, int k0, unsigned short* ldsDst)
{
    unsigned long long ga = (unsigned long long)(uintptr_t)(gA + (size_t)m0 * K + k0);
    unsigned lds = (unsigned)(uintptr_t)ldsDst;   // generic->LDS: low 32 bits are the LDS byte offset
    v4u g0;
    g0[0] = 1u;                                   // D# count = 1 (valid user descriptor)
    g0[1] = lds;                                  // lds_addr
    g0[2] = (unsigned)(ga & 0xFFFFFFFFu);         // global_addr[31:0]
    g0[3] = (unsigned)((ga >> 32) & 0x01FFFFFFu) | (2u << 30);   // global_addr[56:32] | type=2
    v8i g1;
    // data_size=2B (code 1), pad_enable, pad_interval=16 DWORDs (code 3), pad_amount=4 DWORDs (code 3)
    g1[0] = (int)((1u << 16) | (1u << 20) | (3u << 22) | (3u << 25));
    g1[1] = (int)(((unsigned)K & 0xFFFFu) << 16);                              // tensor_dim0 lo16
    g1[2] = (int)((((unsigned)K >> 16) & 0xFFFFu) | (((unsigned)M & 0xFFFFu) << 16)); // dim0 hi | dim1 lo
    g1[3] = (int)((((unsigned)M >> 16) & 0xFFFFu) | (32u << 16));              // dim1 hi | tile_dim0=32
    g1[4] = 128;                                                               // tile_dim1 = 128 rows
    g1[5] = K;                                                                 // tensor_dim0_stride lo32
    g1[6] = 0;
    g1[7] = 0;
    v4i gz; gz[0] = 0; gz[1] = 0; gz[2] = 0; gz[3] = 0;
#if defined(__clang_major__) && (__clang_major__ >= 23)
    v8i gz8; 
#pragma unroll
    for (int i = 0; i < 8; ++i) gz8[i] = 0;
    __builtin_amdgcn_tensor_load_to_lds(g0, g1, gz, gz, gz8, 0);
#else
    __builtin_amdgcn_tensor_load_to_lds(g0, g1, gz, gz, 0);
#endif
}
#endif

// ---- register-batched staging helpers (issue all global loads first) ------
__device__ __forceinline__ void load_B_regs(const float* __restrict__ B, int N, int n0, int k0,
                                            int tid, float4* breg)
{
#pragma unroll
    for (int it = 0; it < 4; ++it) {
        int idx = it * 256 + tid;
        int k = idx >> 5, n = (idx & 31) * 4;
        const float* src = B + (size_t)(k0 + k) * N + n0 + n;
        float4 v;
        if (n0 + n + 4 <= N) {
            v = *(const float4*)src;
        } else {
            v.x = (n0 + n + 0 < N) ? src[0] : 0.f;
            v.y = (n0 + n + 1 < N) ? src[1] : 0.f;
            v.z = (n0 + n + 2 < N) ? src[2] : 0.f;
            v.w = (n0 + n + 3 < N) ? src[3] : 0.f;
        }
        breg[it] = v;
    }
}
__device__ __forceinline__ void store_B_lds(unsigned short (*Bsb)[40], int tid, const float4* breg)
{
#pragma unroll
    for (int it = 0; it < 4; ++it) {
        int idx = it * 256 + tid;
        int k = idx >> 5, n = (idx & 31) * 4;
        Bsb[n + 0][k] = f2bfu(breg[it].x);
        Bsb[n + 1][k] = f2bfu(breg[it].y);
        Bsb[n + 2][k] = f2bfu(breg[it].z);
        Bsb[n + 3][k] = f2bfu(breg[it].w);
    }
}
__device__ __forceinline__ void load_A_regs(const unsigned short* __restrict__ A, int K,
                                            int m0, int k0, int tid, u16x8* areg)
{
#pragma unroll
    for (int it = 0; it < 2; ++it) {
        int idx = it * 256 + tid;
        int row = idx >> 2, cg = idx & 3;
        areg[it] = *(const u16x8*)(A + (size_t)(m0 + row) * K + k0 + cg * 8);
    }
}
__device__ __forceinline__ void store_A_lds(unsigned short (*Asb)[40], int tid, const u16x8* areg)
{
#pragma unroll
    for (int it = 0; it < 2; ++it) {
        int idx = it * 256 + tid;
        int row = idx >> 2, cg = idx & 3;
        *(u16x8*)&Asb[row][cg * 8] = areg[it];
    }
}

// ===========================================================================
// Generic GEMM: C[MxN] = A_bf16[MxK] * B_f32[KxN] (+ addend), bf16 WMMA, f32 acc
// 256 thr = 8 waves, 128x128 tile, K-step 32, double-buffered LDS.
// A tile fetched by the Tensor Data Mover (async, TENSORcnt); B tile converted
// fp32->bf16 through registers, stored to LDS *after* the WMMA block so global
// load latency overlaps matrix math. M % 128 == 0, K % 32 == 0.
// ===========================================================================
__global__ __launch_bounds__(256) void gemm_bf16_kernel(
    const unsigned short* __restrict__ A, const float* __restrict__ B,
    float* __restrict__ C, const float* __restrict__ addend,
    int M, int N, int K)
{
    __shared__ __align__(16) unsigned short As[2][128][40];
    __shared__ __align__(16) unsigned short Bs[2][128][40];
    const int tid = threadIdx.x;
    const int lane = tid & 31, lane16 = lane & 15, kh = lane >> 4;
    const int wave = tid >> 5;
    const int m0 = blockIdx.y * 128, n0 = blockIdx.x * 128;
    const int wm = wave >> 2, wn = wave & 3;   // 2 x 4 wave grid -> 64x32 per wave
    v8f acc[4][2] = {};
    const int nk = K >> 5;
#if HAVE_TDM
    const bool tdmw = (__builtin_amdgcn_readfirstlane(wave) == 0);
#endif

    // prologue: stage tile 0 into buffer 0
    {
        float4 breg[4];
#if HAVE_TDM
        if (tdmw) tdm_load_A_tile(A, M, K, m0, 0, &As[0][0][0]);
#else
        u16x8 areg[2];
        load_A_regs(A, K, m0, 0, tid, areg);
#endif
        load_B_regs(B, N, n0, 0, tid, breg);
#if !HAVE_TDM
        store_A_lds(As[0], tid, areg);
#endif
        store_B_lds(Bs[0], tid, breg);
    }

    for (int kt = 0; kt < nk; ++kt) {
        const int cur = kt & 1;
#if HAVE_TDM
        if (tdmw) __builtin_amdgcn_s_wait_tensorcnt(0);
#endif
        __syncthreads();
        const bool hasNext = (kt + 1 < nk);
        float4 breg[4];
#if !HAVE_TDM
        u16x8 areg[2];
#endif
        if (hasNext) {
            const int k0n = (kt + 1) << 5;
#if HAVE_TDM
            if (tdmw) tdm_load_A_tile(A, M, K, m0, k0n, &As[cur ^ 1][0][0]);
#else
            load_A_regs(A, K, m0, k0n, tid, areg);
#endif
            load_B_regs(B, N, n0, k0n, tid, breg);
            if (kt + 2 < nk) {   // gfx1250 global_prefetch_b8 of tile after next
                int k = tid >> 3, n = (tid & 7) * 16;
                __builtin_prefetch(B + (size_t)(((kt + 2) << 5) + k) * N + n0 + n, 0, 1);
            }
        }
        // compute on current buffer
        v16bf af[4];
#pragma unroll
        for (int mi = 0; mi < 4; ++mi) af[mi] = frag_a(&As[cur][wm * 64 + mi * 16 + lane16][0], 0, kh);
        v16bf bfr[2];
#pragma unroll
        for (int ni = 0; ni < 2; ++ni) bfr[ni] = frag_b(&Bs[cur][wn * 32 + ni * 16 + lane16][0], 0, kh);
#pragma unroll
        for (int mi = 0; mi < 4; ++mi)
#pragma unroll
            for (int ni = 0; ni < 2; ++ni)
                acc[mi][ni] = wmma_bf(af[mi], bfr[ni], acc[mi][ni]);
        // store next tile into the other buffer (visible after next barrier)
        if (hasNext) {
#if !HAVE_TDM
            store_A_lds(As[cur ^ 1], tid, areg);
#endif
            store_B_lds(Bs[cur ^ 1], tid, breg);
        }
    }
    // epilogue: C-layout lane -> (M = v + 8*kh, N = lane16)
#pragma unroll
    for (int mi = 0; mi < 4; ++mi)
#pragma unroll
        for (int ni = 0; ni < 2; ++ni) {
            int row0 = m0 + wm * 64 + mi * 16 + kh * 8;
            int col  = n0 + wn * 32 + ni * 16 + lane16;
            if (col < N) {
#pragma unroll
                for (int v = 0; v < 8; ++v) {
                    size_t o = (size_t)(row0 + v) * N + col;
                    float val = acc[mi][ni][v];
                    if (addend) val += addend[o];
                    C[o] = val;
                }
            }
        }
}

// ===========================================================================
// SSD (Mamba2 chunked scan): one block per (head, batch), 16 chunks of 128.
// All intra-chunk matmuls via WMMA bf16; scan/decays/recurrence in fp32 LDS.
// ===========================================================================
#define SSD_LDS_BYTES (((3 * 128 * LDN + 64 * LDN) * 2) + ((64 * LDN + 4 * 128) * 4))

__global__ __launch_bounds__(256) void ssd_kernel(
    const float* __restrict__ xBC, const float* __restrict__ dtb,
    const float* __restrict__ A_log, const float* __restrict__ Dp,
    float* __restrict__ y)
{
    extern __shared__ __align__(16) char smem[];
    unsigned short* sC  = (unsigned short*)smem;        // [128][LDN] C (later state_decay-scaled)
    unsigned short* sBM = sC  + 128 * LDN;              // [128][LDN] B, then masked M
    unsigned short* sBt = sBM + 128 * LDN;              // [128][LDN] (B*decay_states)^T  [n][l]
    unsigned short* sXt = sBt + 128 * LDN;              // [64][LDN]  (x*dt)^T            [p][l]
    float* sPrev = (float*)(sXt + 64 * LDN);            // [64][LDN]  prev state^T        [p][n]
    float* sScan = sPrev + 64 * LDN;                    // [128] A_cum
    float* sSD   = sScan + 128;                         // [128] exp(A_cum)
    float* sDS   = sSD + 128;                           // [128] exp(A_last - A_cum)
    float* sDT   = sDS + 128;                           // [128] dt

    const int tid = threadIdx.x;
    const int wave = tid >> 5, lane = tid & 31, lane16 = lane & 15, kh = lane >> 4;
    const int h = blockIdx.x, b = blockIdx.y;
    const float Ah = -__expf(A_log[h]);
    const float Dh = Dp[h];

    for (int i = tid; i < 64 * LDN; i += 256) sPrev[i] = 0.f;
    __syncthreads();

    for (int c = 0; c < NCHUNK; ++c) {
        const int t0 = b * LSEQ + c * CHUNKL;
        // --- phase 1: dA, inclusive scan (Hillis-Steele) ---
        if (tid < 128) {
            float dv = dtb[(size_t)(t0 + tid) * NHEADS + h];
            sDT[tid] = dv;
            sScan[tid] = dv * Ah;
        }
        __syncthreads();
        for (int off = 1; off < 128; off <<= 1) {
            float v = 0.f;
            if (tid < 128) v = sScan[tid] + ((tid >= off) ? sScan[tid - off] : 0.f);
            __syncthreads();
            if (tid < 128) sScan[tid] = v;
            __syncthreads();
        }
        const float Alast = sScan[127];
        if (tid < 128) { sSD[tid] = __expf(sScan[tid]); sDS[tid] = __expf(Alast - sScan[tid]); }
        __syncthreads();
        // --- phase 2: stage C, B, (B*decay)^T, (x*dt)^T in LDS as bf16 ---
#pragma unroll 4
        for (int idx = tid; idx < 128 * 128; idx += 256) {
            int l = idx >> 7, n = idx & 127;
            const float* base = xBC + (size_t)(t0 + l) * CONVDIM;
            float Bv = base[DINNER + n];
            float Cv = base[DINNER + DSTATE + n];
            sBM[l * LDN + n] = f2bfu(Bv);
            sC [l * LDN + n] = f2bfu(Cv);
            sBt[n * LDN + l] = f2bfu(Bv * sDS[l]);
        }
#pragma unroll 4
        for (int idx = tid; idx < 128 * 64; idx += 256) {
            int l = idx >> 6, p = idx & 63;
            float xv = xBC[(size_t)(t0 + l) * CONVDIM + h * HEADDIM + p];
            sXt[p * LDN + l] = f2bfu(xv * sDT[l]);
        }
        __syncthreads();
        // --- phase 3: G = C @ B^T (128x128), then M = tril(G) * exp(Acum[l]-Acum[s]) ---
        {
            const int wm = wave >> 2, wn = wave & 3;
            v8f g[4][2] = {};
            for (int k0 = 0; k0 < 128; k0 += 32) {
                v16bf af[4], bfr[2];
#pragma unroll
                for (int mi = 0; mi < 4; ++mi) af[mi] = frag_a(sC + (size_t)(wm * 64 + mi * 16 + lane16) * LDN, k0, kh);
#pragma unroll
                for (int ni = 0; ni < 2; ++ni) bfr[ni] = frag_b(sBM + (size_t)(wn * 32 + ni * 16 + lane16) * LDN, k0, kh);
#pragma unroll
                for (int mi = 0; mi < 4; ++mi)
#pragma unroll
                    for (int ni = 0; ni < 2; ++ni) g[mi][ni] = wmma_bf(af[mi], bfr[ni], g[mi][ni]);
            }
            __syncthreads();   // all reads of sBM (B) done -> safe to overwrite with M
#pragma unroll
            for (int mi = 0; mi < 4; ++mi)
#pragma unroll
                for (int ni = 0; ni < 2; ++ni) {
                    int l0 = wm * 64 + mi * 16 + kh * 8;
                    int s  = wn * 32 + ni * 16 + lane16;
#pragma unroll
                    for (int v = 0; v < 8; ++v) {
                        int l = l0 + v;
                        float val = (s <= l) ? g[mi][ni][v] * __expf(sScan[l] - sScan[s]) : 0.f;
                        sBM[l * LDN + s] = f2bfu(val);
                    }
                }
        }
        // scale C rows by state_decay = exp(A_cum) (for Y_off term)
#pragma unroll 4
        for (int idx = tid; idx < 128 * 128; idx += 256) {
            int l = idx >> 7, n = idx & 127;
            sC[l * LDN + n] = f2bfu(bfu2f(sC[l * LDN + n]) * sSD[l]);
        }
        __syncthreads();
        // --- phase 4: Y = M @ Xdt + (C*sd) @ Prev + D*x  (128x64) ---
        {
            const int wm = wave >> 1, wn = wave & 1;   // 4 x 2 -> 32x32 per wave
            v8f yd[2][2] = {};
            for (int k0 = 0; k0 < 128; k0 += 32) {
                v16bf af[2], bfr[2];
#pragma unroll
                for (int mi = 0; mi < 2; ++mi) af[mi] = frag_a(sBM + (size_t)(wm * 32 + mi * 16 + lane16) * LDN, k0, kh);
#pragma unroll
                for (int ni = 0; ni < 2; ++ni) bfr[ni] = frag_b(sXt + (size_t)(wn * 32 + ni * 16 + lane16) * LDN, k0, kh);
#pragma unroll
                for (int mi = 0; mi < 2; ++mi)
#pragma unroll
                    for (int ni = 0; ni < 2; ++ni) yd[mi][ni] = wmma_bf(af[mi], bfr[ni], yd[mi][ni]);
            }
            for (int k0 = 0; k0 < 128; k0 += 32) {
                v16bf af[2], bfr[2];
#pragma unroll
                for (int mi = 0; mi < 2; ++mi) af[mi] = frag_a(sC + (size_t)(wm * 32 + mi * 16 + lane16) * LDN, k0, kh);
#pragma unroll
                for (int ni = 0; ni < 2; ++ni) bfr[ni] = frag_b_f32(sPrev + (size_t)(wn * 32 + ni * 16 + lane16) * LDN, k0, kh);
#pragma unroll
                for (int mi = 0; mi < 2; ++mi)
#pragma unroll
                    for (int ni = 0; ni < 2; ++ni) yd[mi][ni] = wmma_bf(af[mi], bfr[ni], yd[mi][ni]);
            }
#pragma unroll
            for (int mi = 0; mi < 2; ++mi)
#pragma unroll
                for (int ni = 0; ni < 2; ++ni) {
                    int l0 = wm * 32 + mi * 16 + kh * 8;
                    int p  = wn * 32 + ni * 16 + lane16;
#pragma unroll
                    for (int v = 0; v < 8; ++v) {
                        int l = l0 + v;
                        float xv = xBC[(size_t)(t0 + l) * CONVDIM + h * HEADDIM + p];
                        y[(size_t)(t0 + l) * DINNER + h * HEADDIM + p] = yd[mi][ni][v] + Dh * xv;
                    }
                }
        }
        // --- phase 5: S = (B*decay)^T @ Xdt ; Prev = exp(Alast)*Prev + S ---
        {
            const int wm = wave >> 1, wn = wave & 1;
            v8f st[2][2] = {};
            for (int k0 = 0; k0 < 128; k0 += 32) {
                v16bf af[2], bfr[2];
#pragma unroll
                for (int mi = 0; mi < 2; ++mi) af[mi] = frag_a(sBt + (size_t)(wm * 32 + mi * 16 + lane16) * LDN, k0, kh);
#pragma unroll
                for (int ni = 0; ni < 2; ++ni) bfr[ni] = frag_b(sXt + (size_t)(wn * 32 + ni * 16 + lane16) * LDN, k0, kh);
#pragma unroll
                for (int mi = 0; mi < 2; ++mi)
#pragma unroll
                    for (int ni = 0; ni < 2; ++ni) st[mi][ni] = wmma_bf(af[mi], bfr[ni], st[mi][ni]);
            }
            __syncthreads();   // ensure all phase-4 reads of sPrev complete
            const float eA = __expf(Alast);
#pragma unroll
            for (int mi = 0; mi < 2; ++mi)
#pragma unroll
                for (int ni = 0; ni < 2; ++ni) {
                    int n0r = wm * 32 + mi * 16 + kh * 8;
                    int p   = wn * 32 + ni * 16 + lane16;
#pragma unroll
                    for (int v = 0; v < 8; ++v) {
                        int n = n0r + v;
                        sPrev[p * LDN + n] = sPrev[p * LDN + n] * eA + st[mi][ni][v];
                    }
                }
        }
        __syncthreads();
    }
}

// ===========================================================================
// Elementwise / norm kernels
// ===========================================================================
__global__ __launch_bounds__(256) void rmsnorm_bf16_kernel(
    const float* __restrict__ x, const float* __restrict__ w,
    unsigned short* __restrict__ out, int K)
{
    const int t = blockIdx.x, tid = threadIdx.x;
    const float* row = x + (size_t)t * K;
    __shared__ float red[256];
    float s = 0.f;
    for (int i = tid; i < K; i += 256) { float v = row[i]; s += v * v; }
    red[tid] = s; __syncthreads();
    for (int o = 128; o > 0; o >>= 1) { if (tid < o) red[tid] += red[tid + o]; __syncthreads(); }
    float scale = rsqrtf(red[0] / (float)K + 1e-6f);
    for (int i = tid; i < K; i += 256) out[(size_t)t * K + i] = f2bfu(row[i] * scale * w[i]);
}

__global__ __launch_bounds__(256) void gatenorm_kernel(
    const float* __restrict__ y, const float* __restrict__ zx,
    const float* __restrict__ w, unsigned short* __restrict__ out)
{
    const int t = blockIdx.x, tid = threadIdx.x;
    __shared__ float g[DINNER];
    __shared__ float red[256];
    float s = 0.f;
    for (int i = tid; i < DINNER; i += 256) {
        float z = zx[(size_t)t * INPROJ + i];
        float gv = y[(size_t)t * DINNER + i] * silu_f(z);
        g[i] = gv; s += gv * gv;
    }
    red[tid] = s; __syncthreads();
    for (int o = 128; o > 0; o >>= 1) { if (tid < o) red[tid] += red[tid + o]; __syncthreads(); }
    float scale = rsqrtf(red[0] / (float)DINNER + 1e-6f);
    for (int i = tid; i < DINNER; i += 256) out[(size_t)t * DINNER + i] = f2bfu(g[i] * scale * w[i]);
}

__global__ __launch_bounds__(256) void conv_silu_kernel(
    const float* __restrict__ zx, const float* __restrict__ cw,
    const float* __restrict__ cb, float* __restrict__ out)
{
    size_t i = (size_t)blockIdx.x * 256 + threadIdx.x;
    if (i >= (size_t)T_TOK * CONVDIM) return;
    int ch = (int)(i % CONVDIM);
    int t = (int)(i / CONVDIM);
    int b = t >> 11, l = t & 2047;
    float acc = cb[ch];
#pragma unroll
    for (int j = 0; j < 4; ++j) {
        int li = l - 3 + j;
        if (li >= 0) acc += cw[ch * 4 + j] * zx[(size_t)(b * LSEQ + li) * INPROJ + DINNER + ch];
    }
    out[i] = silu_f(acc);
}

__global__ __launch_bounds__(256) void dt_softplus_kernel(
    const float* __restrict__ zx, const float* __restrict__ dt_bias,
    float* __restrict__ out)
{
    int i = blockIdx.x * 256 + threadIdx.x;
    if (i >= T_TOK * NHEADS) return;
    int h = i & 63, t = i >> 6;
    float x = zx[(size_t)t * INPROJ + (INPROJ - NHEADS) + h] + dt_bias[h];
    out[i] = (x > 20.f) ? x : log1pf(__expf(x));
}

__global__ __launch_bounds__(256) void actmul_kernel(
    const float* __restrict__ g, const float* __restrict__ u,
    unsigned short* __restrict__ out)
{
    size_t i = (size_t)blockIdx.x * 256 + threadIdx.x;
    if (i >= (size_t)T_TOK * INTERM) return;
    out[i] = f2bfu(silu_f(g[i]) * u[i]);
}

// ===========================================================================
extern "C" void kernel_launch(void* const* d_in, const int* in_sizes, int n_in,
                              void* d_out, int out_size, void* d_ws, size_t ws_size,
                              hipStream_t stream)
{
    const float* hidden       = (const float*)d_in[0];
    const float* norm1_w      = (const float*)d_in[1];
    const float* in_proj_w    = (const float*)d_in[2];
    const float* conv_w       = (const float*)d_in[3];
    const float* conv_b       = (const float*)d_in[4];
    const float* dt_bias      = (const float*)d_in[5];
    const float* A_log        = (const float*)d_in[6];
    const float* D_param      = (const float*)d_in[7];
    const float* mixer_norm_w = (const float*)d_in[8];
    const float* out_proj_w   = (const float*)d_in[9];
    const float* norm2_w      = (const float*)d_in[10];
    const float* gate_w       = (const float*)d_in[11];
    const float* up_w         = (const float*)d_in[12];
    const float* down_w       = (const float*)d_in[13];
    float* outp = (float*)d_out;
    (void)in_sizes; (void)n_in; (void)out_size; (void)ws_size;

    char* ws = (char*)d_ws; size_t off = 0;
    auto alloc = [&](size_t bytes) -> void* {
        void* p = ws + off;
        off += (bytes + 255) & ~(size_t)255;
        return p;
    };
    unsigned short* hbf   = (unsigned short*)alloc((size_t)T_TOK * DMODEL * 2);
    float*          zx    = (float*)alloc((size_t)T_TOK * INPROJ * 4);
    float*          xbc   = (float*)alloc((size_t)T_TOK * CONVDIM * 4);
    float*          dtb   = (float*)alloc((size_t)T_TOK * NHEADS * 4);
    float*          ybuf  = (float*)alloc((size_t)T_TOK * DINNER * 4);
    unsigned short* ynbf  = (unsigned short*)alloc((size_t)T_TOK * DINNER * 2);
    float*          h1    = (float*)alloc((size_t)T_TOK * DMODEL * 4);
    unsigned short* h2bf  = (unsigned short*)alloc((size_t)T_TOK * DMODEL * 2);
    float*          gbuf  = (float*)alloc((size_t)T_TOK * INTERM * 4);
    float*          ubuf  = (float*)alloc((size_t)T_TOK * INTERM * 4);
    unsigned short* actbf = (unsigned short*)alloc((size_t)T_TOK * INTERM * 2);

    hipFuncSetAttribute((const void*)ssd_kernel,
                        hipFuncAttributeMaxDynamicSharedMemorySize, SSD_LDS_BYTES);

    // 1) norm1 -> bf16
    rmsnorm_bf16_kernel<<<T_TOK, 256, 0, stream>>>(hidden, norm1_w, hbf, DMODEL);
    // 2) in_proj GEMM 4096x8512x2048
    gemm_bf16_kernel<<<dim3((INPROJ + 127) / 128, T_TOK / 128), 256, 0, stream>>>(
        hbf, in_proj_w, zx, nullptr, T_TOK, INPROJ, DMODEL);
    // 3) depthwise causal conv + SiLU; dt softplus
    conv_silu_kernel<<<(int)(((size_t)T_TOK * CONVDIM + 255) / 256), 256, 0, stream>>>(zx, conv_w, conv_b, xbc);
    dt_softplus_kernel<<<(T_TOK * NHEADS + 255) / 256, 256, 0, stream>>>(zx, dt_bias, dtb);
    // 4) SSD chunked scan (WMMA per-chunk GEMMs)
    ssd_kernel<<<dim3(NHEADS, 2), 256, SSD_LDS_BYTES, stream>>>(xbc, dtb, A_log, D_param, ybuf);
    // 5) gated mixer RMSNorm -> bf16
    gatenorm_kernel<<<T_TOK, 256, 0, stream>>>(ybuf, zx, mixer_norm_w, ynbf);
    // 6) out_proj GEMM + residual
    gemm_bf16_kernel<<<dim3(DMODEL / 128, T_TOK / 128), 256, 0, stream>>>(
        ynbf, out_proj_w, h1, hidden, T_TOK, DMODEL, DINNER);
    // 7) norm2 -> bf16
    rmsnorm_bf16_kernel<<<T_TOK, 256, 0, stream>>>(h1, norm2_w, h2bf, DMODEL);
    // 8) SwiGLU MLP
    gemm_bf16_kernel<<<dim3(INTERM / 128, T_TOK / 128), 256, 0, stream>>>(
        h2bf, gate_w, gbuf, nullptr, T_TOK, INTERM, DMODEL);
    gemm_bf16_kernel<<<dim3(INTERM / 128, T_TOK / 128), 256, 0, stream>>>(
        h2bf, up_w, ubuf, nullptr, T_TOK, INTERM, DMODEL);
    actmul_kernel<<<(int)(((size_t)T_TOK * INTERM + 255) / 256), 256, 0, stream>>>(gbuf, ubuf, actbf);
    gemm_bf16_kernel<<<dim3(DMODEL / 128, T_TOK / 128), 256, 0, stream>>>(
        actbf, down_w, outp, h1, T_TOK, DMODEL, INTERM);
}
